// PatchSVAEv2_82377472737308
// MI455X (gfx1250) — matrix operand order repaired and checked by reading, not compiled, not run on hardware
//
#include <hip/hip_runtime.h>
#include <hip/hip_bf16.h>
#include <math.h>

// ---------------------------------------------------------------------------
// Problem constants (match reference)
// ---------------------------------------------------------------------------
#define BB      32
#define NN      1024
#define NTOK    (BB * NN)          // 32768
#define PD_     48
#define PDPAD   64
#define HH      384
#define DEPTH_  4
#define VV      48
#define DD      4
#define NCROSS_ 2

// output offsets (floats) in d_out
#define U_OFF   0L
#define S_OFF   6291456L           // NTOK*VV*DD
#define SC_OFF  6422528L
#define VT_OFF  6553600L
#define M_OFF   7077888L

// workspace byte offsets
#define WS_H      0L                       // f32 h: NTOK*HH*4       = 50331648
#define WS_ABUF   50331648L                // bf16 A: NTOK*HH*2      = 25165824
#define WS_BBUF   75497472L                // bf16 B: NTOK*HH*2      = 25165824
#define WS_WINT   100663296L               // bf16 [384][64]         = 49152
#define WS_WB1T   100712448L               // bf16 4*[384][384]      = 1179648
#define WS_WB2T   101892096L               // bf16 4*[384][384]      = 1179648
#define WS_WOUTT  103071744L               // bf16 [192][384]        = 147456

typedef __attribute__((ext_vector_type(16))) __bf16 v16bf;
typedef __attribute__((ext_vector_type(8)))  __bf16 v8bf;
typedef __attribute__((ext_vector_type(8)))  float  v8f;
typedef __attribute__((ext_vector_type(4)))  float  v4f;
typedef __attribute__((ext_vector_type(4)))  int    i32x4;

typedef __attribute__((address_space(1))) i32x4 gi32x4;   // global int4
typedef __attribute__((address_space(3))) i32x4 li32x4;   // LDS int4

#if defined(__has_builtin)
#if __has_builtin(__builtin_amdgcn_global_load_async_to_lds_b128)
#define HAVE_ASYNC_LDS 1
#else
#define HAVE_ASYNC_LDS 0
#endif
#else
#define HAVE_ASYNC_LDS 0
#endif

template <int N>
__device__ __forceinline__ void wait_asynccnt() {
#if defined(__has_builtin) && __has_builtin(__builtin_amdgcn_s_wait_asynccnt)
    __builtin_amdgcn_s_wait_asynccnt(N);
#else
    asm volatile("s_wait_asynccnt %0" ::"i"(N) : "memory");
#endif
}

__device__ __forceinline__ unsigned short f2bf(float f) {
    unsigned u = __float_as_uint(f);
    unsigned r = u + 0x7FFFu + ((u >> 16) & 1u);   // round-to-nearest-even
    return (unsigned short)(r >> 16);
}
__device__ __forceinline__ float gelu_exact(float x) {
    return 0.5f * x * (1.0f + erff(x * 0.70710678118654752f));
}

// ---------------------------------------------------------------------------
// Weight transpose + f32 -> bf16 (dst[b][n][kp] = src[b][kp][n], pad K)
// ---------------------------------------------------------------------------
__global__ void k_wtrans(const float* __restrict__ src, unsigned short* __restrict__ dst,
                         int K, int N, int Kpad, int batch) {
    long total = (long)batch * N * Kpad;
    for (long i = (long)blockIdx.x * blockDim.x + threadIdx.x; i < total;
         i += (long)gridDim.x * blockDim.x) {
        int  kp = (int)(i % Kpad);
        long r  = i / Kpad;
        int  n  = (int)(r % N);
        int  b  = (int)(r / N);
        float v = (kp < K) ? src[((long)b * K + kp) * N + n] : 0.0f;
        dst[i] = f2bf(v);
    }
}

// patches [NTOK][48] f32 -> [NTOK][64] bf16 (zero pad)
__global__ void k_padpatches(const float* __restrict__ src, unsigned short* __restrict__ dst) {
    long total = (long)NTOK * PDPAD;
    for (long i = (long)blockIdx.x * blockDim.x + threadIdx.x; i < total;
         i += (long)gridDim.x * blockDim.x) {
        int  kp = (int)(i & (PDPAD - 1));
        long t  = i >> 6;
        float v = (kp < PD_) ? src[t * PD_ + kp] : 0.0f;
        dst[i] = f2bf(v);
    }
}

// f32 -> bf16 elementwise
__global__ void k_f2bf(const float* __restrict__ src, unsigned short* __restrict__ dst, long n) {
    for (long i = (long)blockIdx.x * blockDim.x + threadIdx.x; i < n;
         i += (long)gridDim.x * blockDim.x)
        dst[i] = f2bf(src[i]);
}

// ---------------------------------------------------------------------------
// WMMA bf16 GEMM: out[M,N] = A[M,K] @ W[K,N]  (W stored transposed Wt[n][k])
// Each wave: 16-row stripe, A fragments for the full K in registers.
// B tiles (16 cols x K) are double-buffered in LDS via async-to-LDS copies
// (ASYNCcnt + split barriers) when the toolchain exposes the builtin;
// otherwise direct global loads.
// EPI: 0=gelu->f32  1=gelu->bf16  2=residual f32  3=bias f32
// ---------------------------------------------------------------------------
template <int K, int EPI>
__global__ __launch_bounds__(256) void k_gemm(const unsigned short* __restrict__ A,
                                              const unsigned short* __restrict__ Wt,
                                              const float* __restrict__ bias,
                                              void* __restrict__ outp,
                                              float* __restrict__ res, int N) {
    constexpr int KF = K / 32;
    const int wid  = threadIdx.x >> 5;
    const int lane = threadIdx.x & 31;
    const int row0 = (blockIdx.x * 8 + wid) * 16;
    const int mrow = row0 + (lane & 15);
    const int kb8  = (lane >> 4) * 8;      // unused; A fragments use kb16 like B
    const int kb16 = (lane >> 4) * 16;     // B-fragment K sub-offset per lane half

    // Load all A fragments (contiguous k order, same as B: lane m holds K=kb16..kb16+15)
    v16bf a[KF];
    const unsigned short* Ar = A + (long)mrow * K;
#pragma unroll
    for (int kk = 0; kk < KF; ++kk) {
        v8bf lo = *(const v8bf*)(Ar + kk * 32 + kb16);
        v8bf hi = *(const v8bf*)(Ar + kk * 32 + kb16 + 8);
#pragma unroll
        for (int t = 0; t < 8; ++t) { a[kk][t] = lo[t]; a[kk][8 + t] = hi[t]; }
    }

    const int ncol   = lane & 15;
    const int ntiles = N >> 4;

#if HAVE_ASYNC_LDS
    // ---- async double-buffered B staging in LDS ----
    constexpr int TILE_HALFS = 16 * K;                    // halfs per 16-col tile
    constexpr int TB128      = (TILE_HALFS * 2) / 16;     // 16-byte chunks per tile
    constexpr int LPT        = (TB128 + 255) / 256;       // async issues per thread
    __shared__ __attribute__((aligned(32))) unsigned short sB[2][TILE_HALFS];
    const int tid = threadIdx.x;

    auto stage = [&](int buf, int nt) {
        const unsigned short* base = Wt + (long)nt * 16 * K;  // tile is contiguous
#pragma unroll
        for (int l = 0; l < LPT; ++l) {
            int i = tid + l * 256;
            if (i < TB128) {
                __builtin_amdgcn_global_load_async_to_lds_b128(
                    (gi32x4*)(base + i * 8),
                    (li32x4*)&sB[buf][i * 8], 0, 0);
            }
        }
    };

    stage(0, 0);
    for (int nt = 0; nt < ntiles; ++nt) {
        const bool more = (nt + 1) < ntiles;
        if (more) stage((nt + 1) & 1, nt + 1);
        if (more) wait_asynccnt<LPT>();   // own tile-nt copies done (in-order)
        else      wait_asynccnt<0>();
        __syncthreads();                  // everyone's tile-nt chunks visible

        const unsigned short* Wl = &sB[nt & 1][ncol * K + kb16];
        v8f c = {};
#pragma unroll
        for (int kk = 0; kk < KF; ++kk) {
            v16bf b = *(const v16bf*)(Wl + kk * 32);       // ds_load_b128 x2
            c = __builtin_amdgcn_wmma_f32_16x16x32_bf16(false, a[kk], false, b,
                                                        (short)0, c, false, false);
        }
        __syncthreads();                  // tile consumed before buffer reuse

        const int  n    = nt * 16 + ncol;
        const float bcol = bias[n];
#pragma unroll
        for (int r = 0; r < 8; ++r) {
            const int  row = row0 + ((lane >> 4) << 3) + r;
            const long idx = (long)row * N + n;
            const float x  = c[r] + bcol;
            if (EPI == 0)      ((float*)outp)[idx] = gelu_exact(x);
            else if (EPI == 1) ((unsigned short*)outp)[idx] = f2bf(gelu_exact(x));
            else if (EPI == 2) res[idx] = res[idx] + x;
            else               ((float*)outp)[idx] = x;
        }
    }
#else
    // ---- fallback: direct global B loads ----
    for (int nt = 0; nt < ntiles; ++nt) {
        const int n = nt * 16 + ncol;
        const unsigned short* Wr = Wt + (long)n * K + kb16;
        v8f c = {};
#pragma unroll
        for (int kk = 0; kk < KF; ++kk) {
            v16bf b = *(const v16bf*)(Wr + kk * 32);
            c = __builtin_amdgcn_wmma_f32_16x16x32_bf16(false, a[kk], false, b,
                                                        (short)0, c, false, false);
        }
        const float bcol = bias[n];
#pragma unroll
        for (int r = 0; r < 8; ++r) {
            const int  row = row0 + ((lane >> 4) << 3) + r;
            const long idx = (long)row * N + n;
            const float x  = c[r] + bcol;
            if (EPI == 0)      ((float*)outp)[idx] = gelu_exact(x);
            else if (EPI == 1) ((unsigned short*)outp)[idx] = f2bf(gelu_exact(x));
            else if (EPI == 2) res[idx] = res[idx] + x;
            else               ((float*)outp)[idx] = x;
        }
    }
#endif
}

// ---------------------------------------------------------------------------
// LayerNorm over H=384, one wave per row, output bf16
// ---------------------------------------------------------------------------
__global__ __launch_bounds__(256) void k_ln_bf16(const float* __restrict__ h,
                                                 unsigned short* __restrict__ out) {
    const int row  = blockIdx.x * 8 + (threadIdx.x >> 5);
    const int lane = threadIdx.x & 31;
    const float* hr = h + (long)row * HH;
    float x[12];
    float s = 0.0f;
#pragma unroll
    for (int j = 0; j < 12; ++j) { x[j] = hr[lane + j * 32]; s += x[j]; }
#pragma unroll
    for (int off = 16; off > 0; off >>= 1) s += __shfl_xor(s, off, 32);
    const float mean = s * (1.0f / HH);
    float vs = 0.0f;
#pragma unroll
    for (int j = 0; j < 12; ++j) { float d = x[j] - mean; vs += d * d; }
#pragma unroll
    for (int off = 16; off > 0; off >>= 1) vs += __shfl_xor(vs, off, 32);
    const float inv = rsqrtf(vs * (1.0f / HH) + 1e-5f);
    unsigned short* orow = out + (long)row * HH;
#pragma unroll
    for (int j = 0; j < 12; ++j) orow[lane + j * 32] = f2bf((x[j] - mean) * inv);
}

// ---------------------------------------------------------------------------
// Per-token spectral kernel: normalize M rows, 4x4 Gram, Jacobi eigh,
// emit U, S, Sc(init), Vt; M normalized in place.
// ---------------------------------------------------------------------------
__device__ __forceinline__ void jacobi_rot(float A[4][4], float V[4][4], int p, int q) {
    const float apq = A[p][q];
    float t;
    if (fabsf(apq) < 1e-36f) {
        t = 0.0f;
    } else {
        const float tau = (A[q][q] - A[p][p]) / (2.0f * apq);
        t = copysignf(1.0f, tau) / (fabsf(tau) + sqrtf(1.0f + tau * tau));
    }
    const float c = rsqrtf(1.0f + t * t);
    const float s = t * c;
#pragma unroll
    for (int k = 0; k < 4; ++k) {
        const float akp = A[k][p], akq = A[k][q];
        A[k][p] = c * akp - s * akq;
        A[k][q] = s * akp + c * akq;
    }
#pragma unroll
    for (int k = 0; k < 4; ++k) {
        const float apk = A[p][k], aqk = A[q][k];
        A[p][k] = c * apk - s * aqk;
        A[q][k] = s * apk + c * aqk;
        const float vkp = V[k][p], vkq = V[k][q];
        V[k][p] = c * vkp - s * vkq;
        V[k][q] = s * vkp + c * vkq;
    }
}

__global__ __launch_bounds__(256) void k_spectral(float* __restrict__ dout) {
    const int t = blockIdx.x * blockDim.x + threadIdx.x;   // token
    if (t >= NTOK) return;
    float* Mb = dout + M_OFF + (long)t * (VV * DD);

    float G[4][4];
#pragma unroll
    for (int i = 0; i < 4; ++i)
#pragma unroll
        for (int j = 0; j < 4; ++j) G[i][j] = 0.0f;

    for (int v = 0; v < VV; ++v) {
        v4f m = *(v4f*)(Mb + v * 4);
        const float nrm = sqrtf(m.x * m.x + m.y * m.y + m.z * m.z + m.w * m.w);
        const float inv = 1.0f / fmaxf(nrm, 1e-12f);
        m *= inv;
        *(v4f*)(Mb + v * 4) = m;
        float mv[4] = {m.x, m.y, m.z, m.w};
#pragma unroll
        for (int i = 0; i < 4; ++i)
#pragma unroll
            for (int j = 0; j < 4; ++j) G[i][j] += mv[i] * mv[j];
    }

    float Vm[4][4] = {{1, 0, 0, 0}, {0, 1, 0, 0}, {0, 0, 1, 0}, {0, 0, 0, 1}};
#pragma unroll
    for (int sweep = 0; sweep < 8; ++sweep) {
        jacobi_rot(G, Vm, 0, 1); jacobi_rot(G, Vm, 0, 2); jacobi_rot(G, Vm, 0, 3);
        jacobi_rot(G, Vm, 1, 2); jacobi_rot(G, Vm, 1, 3); jacobi_rot(G, Vm, 2, 3);
    }
    float lam[4] = {G[0][0], G[1][1], G[2][2], G[3][3]};
#define CSWAP(i, j)                                                     \
    if (lam[i] < lam[j]) {                                              \
        float tl = lam[i]; lam[i] = lam[j]; lam[j] = tl;                \
        _Pragma("unroll") for (int k = 0; k < 4; ++k) {                 \
            float tv = Vm[k][i]; Vm[k][i] = Vm[k][j]; Vm[k][j] = tv; }  \
    }
    CSWAP(0, 1) CSWAP(1, 2) CSWAP(2, 3) CSWAP(0, 1) CSWAP(1, 2) CSWAP(0, 1)
#undef CSWAP

    float S[4], Sinv[4];
#pragma unroll
    for (int e = 0; e < 4; ++e) {
        S[e]    = sqrtf(fmaxf(lam[e], 1e-24f));
        Sinv[e] = 1.0f / fmaxf(S[e], 1e-16f);
        dout[S_OFF + (long)t * 4 + e]  = S[e];
        dout[SC_OFF + (long)t * 4 + e] = S[e];           // Sc init = S
    }
#pragma unroll
    for (int d = 0; d < 4; ++d)
#pragma unroll
        for (int e = 0; e < 4; ++e)
            dout[VT_OFF + (long)t * 16 + d * 4 + e] = Vm[e][d];

    float* Ub = dout + U_OFF + (long)t * (VV * DD);
    for (int v = 0; v < VV; ++v) {
        v4f m = *(v4f*)(Mb + v * 4);
        float mv[4] = {m.x, m.y, m.z, m.w};
        v4f u;
#pragma unroll
        for (int e = 0; e < 4; ++e) {
            float acc = 0.0f;
#pragma unroll
            for (int d = 0; d < 4; ++d) acc += mv[d] * Vm[d][e];
            u[e] = acc * Sinv[e];
        }
        *(v4f*)(Ub + v * 4) = u;
    }
}

// ---------------------------------------------------------------------------
// Cross-attention update on Sc (D=4, NH=2, HD=2), one block per batch
// ---------------------------------------------------------------------------
__global__ __launch_bounds__(256) void k_cross(float* __restrict__ Sc,
                                               const float* __restrict__ Wqkv,
                                               const float* __restrict__ bqkv,
                                               const float* __restrict__ Wo,
                                               const float* __restrict__ bo,
                                               const float* __restrict__ alpha_logits,
                                               int iter) {
    __shared__ float s_q[NN * 4];
    __shared__ float s_k[NN * 4];
    __shared__ float s_v[NN * 4];
    const int b   = blockIdx.x;
    const int tid = threadIdx.x;
    float* scB = Sc + (long)b * NN * DD;

    for (int n = tid; n < NN; n += 256) {
        float x[4];
        float m = 0.0f;
#pragma unroll
        for (int d = 0; d < 4; ++d) { x[d] = scB[n * 4 + d]; m += x[d]; }
        m *= 0.25f;
        float var = 0.0f;
#pragma unroll
        for (int d = 0; d < 4; ++d) { float dd = x[d] - m; var += dd * dd; }
        const float inv = rsqrtf(var * 0.25f + 1e-5f);
        float xn[4];
#pragma unroll
        for (int d = 0; d < 4; ++d) xn[d] = (x[d] - m) * inv;
#pragma unroll
        for (int c = 0; c < 12; ++c) {
            float acc = bqkv[iter * 12 + c];
#pragma unroll
            for (int d = 0; d < 4; ++d) acc += xn[d] * Wqkv[iter * 48 + d * 12 + c];
            if (c < 4)      s_q[n * 4 + c]       = acc;
            else if (c < 8) s_k[n * 4 + (c - 4)] = acc;
            else            s_v[n * 4 + (c - 8)] = acc;
        }
    }
    __syncthreads();

    const float scale = 0.70710678118654752f;   // HD^-0.5
    for (int n = tid; n < NN; n += 256) {
        float out[4];
#pragma unroll
        for (int h = 0; h < 2; ++h) {
            const float q0 = s_q[n * 4 + h * 2 + 0];
            const float q1 = s_q[n * 4 + h * 2 + 1];
            float mx = -INFINITY, den = 0.0f, a0 = 0.0f, a1 = 0.0f;
            for (int mtok = 0; mtok < NN; ++mtok) {
                const float sc = (q0 * s_k[mtok * 4 + h * 2 + 0] +
                                  q1 * s_k[mtok * 4 + h * 2 + 1]) * scale;
                const float nm   = fmaxf(mx, sc);
                const float corr = __expf(mx - nm);
                const float e    = __expf(sc - nm);
                den = den * corr + e;
                a0  = a0 * corr + e * s_v[mtok * 4 + h * 2 + 0];
                a1  = a1 * corr + e * s_v[mtok * 4 + h * 2 + 1];
                mx  = nm;
            }
            out[h * 2 + 0] = a0 / den;
            out[h * 2 + 1] = a1 / den;
        }
#pragma unroll
        for (int d = 0; d < 4; ++d) {
            float g = bo[iter * 4 + d];
#pragma unroll
            for (int e = 0; e < 4; ++e) g += out[e] * Wo[iter * 16 + e * 4 + d];
            g = tanhf(g);
            const float alpha = 0.2f / (1.0f + __expf(-alpha_logits[iter * 4 + d]));
            scB[n * 4 + d] = scB[n * 4 + d] * (1.0f + alpha * g);
        }
    }
}

// ---------------------------------------------------------------------------
// Launch
// ---------------------------------------------------------------------------
extern "C" void kernel_launch(void* const* d_in, const int* in_sizes, int n_in,
                              void* d_out, int out_size, void* d_ws, size_t ws_size,
                              hipStream_t stream) {
    const float* patches  = (const float*)d_in[0];
    const float* W_in     = (const float*)d_in[1];
    const float* b_in     = (const float*)d_in[2];
    const float* Wb1      = (const float*)d_in[3];
    const float* bb1      = (const float*)d_in[4];
    const float* Wb2      = (const float*)d_in[5];
    const float* bb2      = (const float*)d_in[6];
    const float* W_out    = (const float*)d_in[7];
    const float* b_out    = (const float*)d_in[8];
    const float* Wqkv     = (const float*)d_in[9];
    const float* bqkv     = (const float*)d_in[10];
    const float* Wo       = (const float*)d_in[11];
    const float* bo       = (const float*)d_in[12];
    const float* alpha    = (const float*)d_in[13];

    char*  ws   = (char*)d_ws;
    float* out  = (float*)d_out;
    float*          h     = (float*)(ws + WS_H);
    unsigned short* abuf  = (unsigned short*)(ws + WS_ABUF);
    unsigned short* bbuf  = (unsigned short*)(ws + WS_BBUF);
    unsigned short* WinT  = (unsigned short*)(ws + WS_WINT);
    unsigned short* Wb1T  = (unsigned short*)(ws + WS_WB1T);
    unsigned short* Wb2T  = (unsigned short*)(ws + WS_WB2T);
    unsigned short* WoutT = (unsigned short*)(ws + WS_WOUTT);

    // weights -> transposed bf16
    k_wtrans<<<96,   256, 0, stream>>>(W_in,  WinT,  PD_, HH, PDPAD, 1);
    k_wtrans<<<2048, 256, 0, stream>>>(Wb1,   Wb1T,  HH,  HH, HH,    DEPTH_);
    k_wtrans<<<2048, 256, 0, stream>>>(Wb2,   Wb2T,  HH,  HH, HH,    DEPTH_);
    k_wtrans<<<288,  256, 0, stream>>>(W_out, WoutT, HH,  192, HH,   1);

    // input projection: gelu(patches @ W_in + b_in) -> h (f32)
    k_padpatches<<<2048, 256, 0, stream>>>(patches, abuf);
    k_gemm<PDPAD, 0><<<NTOK / 128, 256, 0, stream>>>(abuf, WinT, b_in, (void*)h, nullptr, HH);

    // residual MLP blocks
    for (int d = 0; d < DEPTH_; ++d) {
        k_ln_bf16<<<NTOK / 8, 256, 0, stream>>>(h, abuf);
        k_gemm<HH, 1><<<NTOK / 128, 256, 0, stream>>>(abuf, Wb1T + (long)d * HH * HH,
                                                      bb1 + d * HH, (void*)bbuf, nullptr, HH);
        k_gemm<HH, 2><<<NTOK / 128, 256, 0, stream>>>(bbuf, Wb2T + (long)d * HH * HH,
                                                      bb2 + d * HH, (void*)h, h, HH);
    }

    // output projection -> raw M directly into d_out M region
    k_f2bf<<<4096, 256, 0, stream>>>(h, abuf, (long)NTOK * HH);
    k_gemm<HH, 3><<<NTOK / 128, 256, 0, stream>>>(abuf, WoutT, b_out,
                                                  (void*)(out + M_OFF), nullptr, 192);

    // spectral decomposition per token (U, S, Sc-init, Vt, normalized M)
    k_spectral<<<NTOK / 256, 256, 0, stream>>>(out);

    // cross-attention refinements of Sc
    for (int i = 0; i < NCROSS_; ++i)
        k_cross<<<BB, 256, 0, stream>>>(out + SC_OFF, Wqkv, bqkv, Wo, bo, alpha, i);
}
